// TrorYongOCRModel_37495064494664
// MI455X (gfx1250) — compile-verified, matrix operand order checked by reading, not run on hardware
//
#include <hip/hip_runtime.h>
#include <hip/hip_bf16.h>
#include <math.h>

// ---------------- model constants ----------------
constexpr int kD   = 1024;
constexpr int kH   = 16;
constexpr int kHD  = 64;
constexpr int kNB  = 7;
constexpr int kFFB = 2048;
constexpr int kFFD = 4096;
constexpr int kV   = 8000;
constexpr int kB   = 32;
constexpr int kL   = 256;
constexpr int kNP  = 16;
constexpr int kS   = 272;
constexpr int kTE  = kB * kNP;   // 512
constexpr int kTD  = kB * kL;    // 8192
constexpr int kTM  = kB * kS;    // 8704

typedef __attribute__((ext_vector_type(16))) _Float16 v16h;
typedef __attribute__((ext_vector_type(8)))  _Float16 v8h;
typedef __attribute__((ext_vector_type(8)))  float    v8f;

// LDS row stride in halves: 80 bytes -> rows 16B-aligned for ds_load_b128,
// 20-word bank stride (20*r mod 64 distinct for r=0..15) -> conflict-free frag reads.
constexpr int kLdsStride = 40;

// ---------------- f32 -> f16 bulk convert (n multiple of 2048) ----------------
__global__ void cvt_f16_kernel(const float* __restrict__ in, _Float16* __restrict__ out,
                               long n)
{
  long i = ((long)blockIdx.x * blockDim.x + threadIdx.x) * 8;
  if (i >= n) return;
  const float4* p = (const float4*)(in + i);
  float4 x0 = p[0], x1 = p[1];
  v8h o;
  o[0] = (_Float16)x0.x; o[1] = (_Float16)x0.y; o[2] = (_Float16)x0.z; o[3] = (_Float16)x0.w;
  o[4] = (_Float16)x1.x; o[5] = (_Float16)x1.y; o[6] = (_Float16)x1.z; o[7] = (_Float16)x1.w;
  *(v8h*)(out + i) = o;
}

// ---------------- tiled WMMA GEMM: C[M,N] = A[M,K] @ W[N,K]^T (+bias)(+act)(+res)
// A, W f16. Block tile 256(M) x 64(N), K-step 32, 8 waves.
// Wave = 64x32 patch = 4x2 WMMA tiles -> 8 v_wmma per K-step vs 12 ds_load_b128.
// Requires: M % 256 == 0, N % 64 == 0, K % 32 == 0 (true for all calls here).
__device__ __forceinline__ v16h frag_load(const _Float16* lds, int rowbase, int hi) {
  union { v16h v; uint4 u[2]; } f;
  f.u[0] = *(const uint4*)(lds + rowbase + hi * 8);
  f.u[1] = *(const uint4*)(lds + rowbase + 16 + hi * 8);
  return f.v;
}

template <int ACT, bool RES>
__global__ __launch_bounds__(256) void gemm_wmma_kernel(
    const _Float16* __restrict__ A, const _Float16* __restrict__ W,
    const float* __restrict__ bias, const float* __restrict__ residual,
    float* __restrict__ C, int M, int N, int K)
{
  __shared__ _Float16 sA[256 * kLdsStride];   // 20.0 KB
  __shared__ _Float16 sB[64 * kLdsStride];    //  5.0 KB

  const int tilesN = N >> 6;
  const int tm = blockIdx.x / tilesN;
  const int tn = blockIdx.x - tm * tilesN;

  const int tid  = threadIdx.x;
  const int wave = tid >> 5;
  const int lane = tid & 31;
  const int r  = lane & 15;
  const int hi = lane >> 4;
  const int wm = wave >> 1;       // 0..3 -> 64-row band
  const int wn = wave & 1;        // 0..1 -> 32-col band

  // cooperative loads: A row per thread (32 halves), B quarter-row per thread
  const _Float16* ga = A + (size_t)(tm * 256 + tid) * (size_t)K;
  const _Float16* gb = W + (size_t)(tn * 64 + (tid >> 2)) * (size_t)K + (tid & 3) * 8;
  _Float16* la = sA + tid * kLdsStride;
  _Float16* lb = sB + (tid >> 2) * kLdsStride + (tid & 3) * 8;

  v8f acc[4][2] = {};

  for (int k0 = 0; k0 < K; k0 += 32) {
    uint4 a0 = *(const uint4*)(ga);
    uint4 a1 = *(const uint4*)(ga + 8);
    uint4 a2 = *(const uint4*)(ga + 16);
    uint4 a3 = *(const uint4*)(ga + 24);
    uint4 b0 = *(const uint4*)(gb);
    if (k0 + 32 < K) {                        // gfx1250 global_prefetch_b8
      __builtin_prefetch(ga + 32, 0, 1);
      __builtin_prefetch(gb + 32, 0, 1);
    }
    __syncthreads();                          // prior iteration's LDS reads done
    *(uint4*)(la)      = a0;
    *(uint4*)(la + 8)  = a1;
    *(uint4*)(la + 16) = a2;
    *(uint4*)(la + 24) = a3;
    *(uint4*)(lb)      = b0;
    __syncthreads();

    v16h af[4], bf[2];
#pragma unroll
    for (int sm = 0; sm < 4; ++sm)
      af[sm] = frag_load(sA, (wm * 64 + sm * 16 + r) * kLdsStride, hi);
#pragma unroll
    for (int sn = 0; sn < 2; ++sn)
      bf[sn] = frag_load(sB, (wn * 32 + sn * 16 + r) * kLdsStride, hi);
#pragma unroll
    for (int sm = 0; sm < 4; ++sm)
#pragma unroll
      for (int sn = 0; sn < 2; ++sn)
        acc[sm][sn] = __builtin_amdgcn_wmma_f32_16x16x32_f16(
            false, af[sm], false, bf[sn], (short)0, acc[sm][sn], false, false);

    ga += 32; gb += 32;
  }

#pragma unroll
  for (int sn = 0; sn < 2; ++sn) {
    const int n = tn * 64 + wn * 32 + sn * 16 + r;
    const float bv = bias ? bias[n] : 0.0f;
#pragma unroll
    for (int sm = 0; sm < 4; ++sm) {
#pragma unroll
      for (int j = 0; j < 8; ++j) {
        const int m = tm * 256 + wm * 64 + sm * 16 + hi * 8 + j;
        float v = acc[sm][sn][j] + bv;
        if (ACT == 1) v = (v > 0.0f) ? v * v : 0.0f;
        if (RES) v += residual[(size_t)m * N + n];
        C[(size_t)m * N + n] = v;
      }
    }
  }
}

// ---------------- RoPE tables ----------------
__global__ void rope_init_kernel(float* __restrict__ cosT, float* __restrict__ sinT) {
  int pos = blockIdx.x;
  int d2  = threadIdx.x;
  float inv = powf(10000.0f, -(2.0f * (float)d2) / (float)kHD);
  float ang = (float)pos * inv;
  cosT[pos * 32 + d2] = cosf(ang);
  sinT[pos * 32 + d2] = sinf(ang);
}

// ---------------- patch extraction ----------------
__global__ void patch_kernel(const float* __restrict__ img, float* __restrict__ P) {
  int row = blockIdx.x;
  int b = row >> 4;
  int p = row & 15;
  int py = p >> 3, px = p & 7;
  for (int e = threadIdx.x; e < 768; e += blockDim.x) {
    int cch = e >> 8;
    int rem = e & 255;
    int i = rem >> 4, j = rem & 15;
    int src = ((b * 3 + cch) * 32 + (py * 16 + i)) * 128 + (px * 16 + j);
    P[(size_t)row * 768 + e] = img[src];
  }
}

// ---------------- derf norm ----------------
__global__ void derf_kernel(const float* __restrict__ x, const float* __restrict__ a,
                            const float* __restrict__ g, const float* __restrict__ bb,
                            float* __restrict__ y, int rows)
{
  int row = blockIdx.x;
  if (row >= rows) return;
  float av = a[0];
  for (int c = threadIdx.x; c < kD; c += blockDim.x) {
    float xv = x[(size_t)row * kD + c];
    y[(size_t)row * kD + c] = g[c] * erff(av * xv) + bb[c];
  }
}

// ---------------- build decoder memory ----------------
__global__ void build_mem_kernel(const float* __restrict__ img_emb,
                                 const float* __restrict__ hn_q,
                                 const float* __restrict__ a, const float* __restrict__ g,
                                 const float* __restrict__ bb,
                                 float* __restrict__ mem)
{
  int row = blockIdx.x;
  int b = row / kS;
  int p = row - b * kS;
  float av = a[0];
  if (p < kNP) {
    const float* src = img_emb + (size_t)(b * kNP + p) * kD;
    for (int c = threadIdx.x; c < kD; c += blockDim.x)
      mem[(size_t)row * kD + c] = g[c] * erff(av * src[c]) + bb[c];
  } else {
    const float* src = hn_q + (size_t)(b * kL + (p - kNP)) * kD;
    for (int c = threadIdx.x; c < kD; c += blockDim.x)
      mem[(size_t)row * kD + c] = src[c];
  }
}

// ---------------- embedding gather ----------------
__global__ void gather_kernel(const int* __restrict__ x, const float* __restrict__ tok,
                              float* __restrict__ out)
{
  int row = blockIdx.x;
  int id = x[row];
  for (int c = threadIdx.x; c < kD; c += blockDim.x)
    out[(size_t)row * kD + c] = tok[(size_t)id * kD + c];
}

// ---------------- per-head RMS + RoPE in place ----------------
__global__ void qk_prep_kernel(float* __restrict__ buf,
                               const float* __restrict__ cosT, const float* __restrict__ sinT,
                               int seqlen, int posoff)
{
  int row = blockIdx.x;
  int h   = blockIdx.y;
  int pos = (row % seqlen) + posoff;
  float* p = buf + (size_t)row * kD + h * kHD;
  int l = threadIdx.x;
  float x1 = p[l];
  float x2 = p[l + 32];
  float ss = x1 * x1 + x2 * x2;
#pragma unroll
  for (int m = 16; m > 0; m >>= 1) ss += __shfl_xor(ss, m, 32);
  float scale = rsqrtf(ss / (float)kHD + 1e-6f);
  x1 *= scale; x2 *= scale;
  float cv = cosT[pos * 32 + l];
  float sv = sinT[pos * 32 + l];
  p[l]      = x1 * cv - x2 * sv;
  p[l + 32] = x1 * sv + x2 * cv;
}

// ---------------- attention (flash-style, thread per query) ----------------
__global__ __launch_bounds__(64) void attention_kernel(
    const float* __restrict__ Q, const float* __restrict__ K,
    const float* __restrict__ Vb, float* __restrict__ O,
    int Lq, int Skv, int causal, int vn_last)
{
  int i = blockIdx.x * 64 + threadIdx.x;
  if (i >= Lq) return;
  int h = blockIdx.y;
  int b = blockIdx.z;
  const float scale = 0.125f;

  const float* qrow = Q + (size_t)(b * Lq + i) * kD + h * kHD;
  float q[kHD];
#pragma unroll
  for (int d = 0; d < kHD; ++d) q[d] = qrow[d];

  int smax = causal ? (Skv - Lq + i) : (Skv - 1);
  float m = -3.0e38f, lsum = 0.0f;
  float o[kHD];
#pragma unroll
  for (int d = 0; d < kHD; ++d) o[d] = 0.0f;

  for (int s = 0; s <= smax; ++s) {
    const float4* k4 = (const float4*)(K + (size_t)(b * Skv + s) * kD + h * kHD);
    float dot = 0.0f;
#pragma unroll
    for (int dd = 0; dd < 16; ++dd) {
      float4 kk = k4[dd];
      dot += q[4 * dd + 0] * kk.x + q[4 * dd + 1] * kk.y +
             q[4 * dd + 2] * kk.z + q[4 * dd + 3] * kk.w;
    }
    float sc = dot * scale;
    float nm = fmaxf(m, sc);
    float f  = __expf(m - nm);
    float p  = __expf(sc - nm);
    lsum = lsum * f + p;
    const float4* v4 = (const float4*)(Vb + (size_t)(b * Skv + s) * kD + h * kHD);
#pragma unroll
    for (int dd = 0; dd < 16; ++dd) {
      float4 vv = v4[dd];
      o[4 * dd + 0] = o[4 * dd + 0] * f + p * vv.x;
      o[4 * dd + 1] = o[4 * dd + 1] * f + p * vv.y;
      o[4 * dd + 2] = o[4 * dd + 2] * f + p * vv.z;
      o[4 * dd + 3] = o[4 * dd + 3] * f + p * vv.w;
    }
    m = nm;
  }
  float inv = 1.0f / lsum;
#pragma unroll
  for (int d = 0; d < kHD; ++d) o[d] *= inv;

  int vidx = vn_last ? (Skv - 1) : i;
  const float* vr = Vb + (size_t)(b * Skv + vidx) * kD + h * kHD;
  float nrm = 0.0f;
#pragma unroll
  for (int d = 0; d < kHD; ++d) nrm += vr[d] * vr[d];
  nrm = fmaxf(sqrtf(nrm), 1e-9f);
  float proj = 0.0f;
#pragma unroll
  for (int d = 0; d < kHD; ++d) proj += o[d] * (vr[d] / nrm);
  float* orow = O + (size_t)(b * Lq + i) * kD + h * kHD;
#pragma unroll
  for (int d = 0; d < kHD; ++d) orow[d] = o[d] - proj * (vr[d] / nrm);
}

// ---------------- RMS norm ----------------
__global__ void rms_kernel(const float* __restrict__ x, float* __restrict__ y, int rows) {
  __shared__ float red[256];
  int row = blockIdx.x;
  if (row >= rows) return;
  float ss = 0.0f;
  for (int c = threadIdx.x; c < kD; c += blockDim.x) {
    float v = x[(size_t)row * kD + c];
    ss += v * v;
  }
  red[threadIdx.x] = ss; __syncthreads();
  for (int s = 128; s > 0; s >>= 1) {
    if (threadIdx.x < s) red[threadIdx.x] += red[threadIdx.x + s];
    __syncthreads();
  }
  float scale = rsqrtf(red[0] / (float)kD + 1e-6f);
  for (int c = threadIdx.x; c < kD; c += blockDim.x)
    y[(size_t)row * kD + c] = x[(size_t)row * kD + c] * scale;
}

// ---------------- loss ----------------
__global__ void loss_rows_kernel(const float* __restrict__ logits,
                                 const int* __restrict__ targets,
                                 float* __restrict__ nll, float* __restrict__ valid)
{
  __shared__ float red[256];
  int r = blockIdx.x;
  const float* row = logits + (size_t)r * kV;
  int t = threadIdx.x;
  float m = -3.0e38f;
  for (int c = t; c < kV; c += 256) m = fmaxf(m, row[c]);
  red[t] = m; __syncthreads();
  for (int s = 128; s > 0; s >>= 1) { if (t < s) red[t] = fmaxf(red[t], red[t + s]); __syncthreads(); }
  m = red[0]; __syncthreads();
  float sum = 0.0f;
  for (int c = t; c < kV; c += 256) sum += expf(row[c] - m);
  red[t] = sum; __syncthreads();
  for (int s = 128; s > 0; s >>= 1) { if (t < s) red[t] += red[t + s]; __syncthreads(); }
  if (t == 0) {
    float lse = m + logf(red[0]);
    int tgt = targets[r];
    float ok = (tgt != 0) ? 1.0f : 0.0f;
    nll[r]   = ok * (lse - row[tgt]);
    valid[r] = ok;
  }
}

__global__ void loss_reduce_kernel(const float* __restrict__ nll,
                                   const float* __restrict__ valid,
                                   float* __restrict__ loss_out)
{
  __shared__ float rn[256], rv[256];
  int t = threadIdx.x;
  float sn = 0.0f, sv = 0.0f;
  for (int r = t; r < kTD; r += 256) { sn += nll[r]; sv += valid[r]; }
  rn[t] = sn; rv[t] = sv; __syncthreads();
  for (int s = 128; s > 0; s >>= 1) {
    if (t < s) { rn[t] += rn[t + s]; rv[t] += rv[t + s]; }
    __syncthreads();
  }
  if (t == 0) loss_out[0] = rn[0] / fmaxf(rv[0], 1.0f);
}

// ---------------- host helpers ----------------
static void cvt(const float* in, _Float16* out, size_t n, hipStream_t s) {
  cvt_f16_kernel<<<dim3((unsigned)(n / 2048)), dim3(256), 0, s>>>(in, out, (long)n);
}
static void gemm(const _Float16* A, const _Float16* W, const float* bias, const float* res,
                 float* C, int M, int N, int K, int act, hipStream_t s)
{
  dim3 grid((M / 256) * (N / 64)), blk(256);
  if (res) {
    if (act) gemm_wmma_kernel<1, true><<<grid, blk, 0, s>>>(A, W, bias, res, C, M, N, K);
    else     gemm_wmma_kernel<0, true><<<grid, blk, 0, s>>>(A, W, bias, res, C, M, N, K);
  } else {
    if (act) gemm_wmma_kernel<1, false><<<grid, blk, 0, s>>>(A, W, bias, res, C, M, N, K);
    else     gemm_wmma_kernel<0, false><<<grid, blk, 0, s>>>(A, W, bias, res, C, M, N, K);
  }
}

extern "C" void kernel_launch(void* const* d_in, const int* in_sizes, int n_in,
                              void* d_out, int out_size, void* d_ws, size_t ws_size,
                              hipStream_t stream)
{
  // setup_inputs order: img, x, targets, params (tree-flattened, keys sorted)
  const float* img     = (const float*)d_in[0];
  const int*   xids    = (const int*)d_in[1];
  const int*   targets = (const int*)d_in[2];
  const float* bW1 = (const float*)d_in[3];
  const float* bW2 = (const float*)d_in[4];
  const float* bWk = (const float*)d_in[5];
  const float* bWo = (const float*)d_in[6];
  const float* bWq = (const float*)d_in[7];
  const float* bWv = (const float*)d_in[8];
  const float* bb1 = (const float*)d_in[9];
  const float* bb2 = (const float*)d_in[10];
  const float* bbk = (const float*)d_in[11];
  const float* bbo = (const float*)d_in[12];
  const float* bbq = (const float*)d_in[13];
  const float* bbv = (const float*)d_in[14];
  const float* bna1 = (const float*)d_in[15];
  const float* bna2 = (const float*)d_in[16];
  const float* bnb1 = (const float*)d_in[17];
  const float* bnb2 = (const float*)d_in[18];
  const float* bng1 = (const float*)d_in[19];
  const float* bng2 = (const float*)d_in[20];
  const float* dW1 = (const float*)d_in[21];
  const float* dW2 = (const float*)d_in[22];
  const float* dWk = (const float*)d_in[23];
  const float* dWo = (const float*)d_in[24];
  const float* dWq = (const float*)d_in[25];
  const float* dWv = (const float*)d_in[26];
  const float* db1 = (const float*)d_in[27];
  const float* db2 = (const float*)d_in[28];
  const float* dbk = (const float*)d_in[29];
  const float* dbo = (const float*)d_in[30];
  const float* dbq = (const float*)d_in[31];
  const float* dbv = (const float*)d_in[32];
  const float* dna1 = (const float*)d_in[33];
  const float* dna2 = (const float*)d_in[34];
  const float* dnb1 = (const float*)d_in[35];
  const float* dnb2 = (const float*)d_in[36];
  const float* dng1 = (const float*)d_in[37];
  const float* dng2 = (const float*)d_in[38];
  const float* lmW  = (const float*)d_in[39];
  const float* lmb  = (const float*)d_in[40];
  const float* patchW = (const float*)d_in[41];
  const float* tokW   = (const float*)d_in[42];

  float* logits = (float*)d_out;
  float* loss   = logits + (size_t)kTD * kV;

  float* ws = (float*)d_ws;
  size_t off = 0;
  float* cosT     = ws + off; off += (size_t)kS * 32;
  float* sinT     = ws + off; off += (size_t)kS * 32;
  float* patches  = ws + off; off += (size_t)kTE * 768;
  float* img_emb  = ws + off; off += (size_t)kTE * kD;
  float* q_buf    = ws + off; off += (size_t)kTD * kD;
  float* k_buf    = ws + off; off += (size_t)kTM * kD;
  float* v_buf    = ws + off; off += (size_t)kTM * kD;
  float* attn_buf = ws + off; off += (size_t)kTD * kD;
  float* hn_buf   = ws + off; off += (size_t)kTD * kD;
  float* mem_buf  = ws + off; off += (size_t)kTM * kD;
  float* mlp_buf  = ws + off; off += (size_t)kTD * kFFD;
  float* query_h  = ws + off; off += (size_t)kTD * kD;
  float* nll_buf  = ws + off; off += kTD;
  float* val_buf  = ws + off; off += kTD;
  _Float16* a_h = (_Float16*)(ws + off); off += ((size_t)kTD * kFFD) / 2;
  _Float16* w_h = (_Float16*)(ws + off); off += ((size_t)kV * kD) / 2;

  // ---- rope + patch embedding ----
  rope_init_kernel<<<dim3(kS), dim3(32), 0, stream>>>(cosT, sinT);
  patch_kernel<<<dim3(kTE), dim3(256), 0, stream>>>(img, patches);
  cvt(patches, a_h, (size_t)kTE * 768, stream);
  cvt(patchW,  w_h, (size_t)kD * 768, stream);
  gemm(a_h, w_h, nullptr, nullptr, img_emb, kTE, kD, 768, 0, stream);

  // ---- encoder blocks ----
  for (int blk = 0; blk < kNB; ++blk) {
    const size_t wD  = (size_t)blk * kD * kD;
    const size_t w1o = (size_t)blk * kFFB * kD;
    derf_kernel<<<dim3(kTE), dim3(256), 0, stream>>>(
        img_emb, bna1 + blk, bng1 + (size_t)blk * kD, bnb1 + (size_t)blk * kD, hn_buf, kTE);
    cvt(hn_buf, a_h, (size_t)kTE * kD, stream);
    cvt(bWq + wD, w_h, (size_t)kD * kD, stream);
    gemm(a_h, w_h, bbq + (size_t)blk * kD, nullptr, q_buf, kTE, kD, kD, 0, stream);
    cvt(bWk + wD, w_h, (size_t)kD * kD, stream);
    gemm(a_h, w_h, bbk + (size_t)blk * kD, nullptr, k_buf, kTE, kD, kD, 0, stream);
    cvt(bWv + wD, w_h, (size_t)kD * kD, stream);
    gemm(a_h, w_h, bbv + (size_t)blk * kD, nullptr, v_buf, kTE, kD, kD, 0, stream);
    qk_prep_kernel<<<dim3(kTE, kH), dim3(32), 0, stream>>>(q_buf, cosT, sinT, kNP, 0);
    qk_prep_kernel<<<dim3(kTE, kH), dim3(32), 0, stream>>>(k_buf, cosT, sinT, kNP, 0);
    attention_kernel<<<dim3(1, kH, kB), dim3(64), 0, stream>>>(
        q_buf, k_buf, v_buf, attn_buf, kNP, kNP, 0, 0);
    cvt(attn_buf, a_h, (size_t)kTE * kD, stream);
    cvt(bWo + wD, w_h, (size_t)kD * kD, stream);
    gemm(a_h, w_h, bbo + (size_t)blk * kD, img_emb, img_emb, kTE, kD, kD, 0, stream);
    derf_kernel<<<dim3(kTE), dim3(256), 0, stream>>>(
        img_emb, bna2 + blk, bng2 + (size_t)blk * kD, bnb2 + (size_t)blk * kD, hn_buf, kTE);
    cvt(hn_buf, a_h, (size_t)kTE * kD, stream);
    cvt(bW1 + w1o, w_h, (size_t)kFFB * kD, stream);
    gemm(a_h, w_h, bb1 + (size_t)blk * kFFB, nullptr, mlp_buf, kTE, kFFB, kD, 1, stream);
    cvt(mlp_buf, a_h, (size_t)kTE * kFFB, stream);
    cvt(bW2 + w1o, w_h, (size_t)kD * kFFB, stream);
    gemm(a_h, w_h, bb2 + (size_t)blk * kD, img_emb, img_emb, kTE, kD, kFFB, 0, stream);
  }

  // ---- decoder block ----
  gather_kernel<<<dim3(kTD), dim3(256), 0, stream>>>(xids, tokW, query_h);
  derf_kernel<<<dim3(kTD), dim3(256), 0, stream>>>(query_h, dna1, dng1, dnb1, hn_buf, kTD);
  build_mem_kernel<<<dim3(kTM), dim3(256), 0, stream>>>(img_emb, hn_buf, dna1, dng1, dnb1, mem_buf);
  cvt(hn_buf, a_h, (size_t)kTD * kD, stream);
  cvt(dWq, w_h, (size_t)kD * kD, stream);
  gemm(a_h, w_h, dbq, nullptr, q_buf, kTD, kD, kD, 0, stream);
  cvt(mem_buf, a_h, (size_t)kTM * kD, stream);
  cvt(dWk, w_h, (size_t)kD * kD, stream);
  gemm(a_h, w_h, dbk, nullptr, k_buf, kTM, kD, kD, 0, stream);
  cvt(dWv, w_h, (size_t)kD * kD, stream);
  gemm(a_h, w_h, dbv, nullptr, v_buf, kTM, kD, kD, 0, stream);
  qk_prep_kernel<<<dim3(kTD, kH), dim3(32), 0, stream>>>(q_buf, cosT, sinT, kL, kNP);
  qk_prep_kernel<<<dim3(kTM, kH), dim3(32), 0, stream>>>(k_buf, cosT, sinT, kS, 0);
  attention_kernel<<<dim3(kL / 64, kH, kB), dim3(64), 0, stream>>>(
      q_buf, k_buf, v_buf, attn_buf, kL, kS, 1, 1);
  cvt(attn_buf, a_h, (size_t)kTD * kD, stream);
  cvt(dWo, w_h, (size_t)kD * kD, stream);
  gemm(a_h, w_h, dbo, query_h, query_h, kTD, kD, kD, 0, stream);
  derf_kernel<<<dim3(kTD), dim3(256), 0, stream>>>(query_h, dna2, dng2, dnb2, hn_buf, kTD);
  cvt(hn_buf, a_h, (size_t)kTD * kD, stream);
  cvt(dW1, w_h, (size_t)kFFD * kD, stream);
  gemm(a_h, w_h, db1, nullptr, mlp_buf, kTD, kFFD, kD, 1, stream);
  cvt(mlp_buf, a_h, (size_t)kTD * kFFD, stream);
  cvt(dW2, w_h, (size_t)kD * kFFD, stream);
  gemm(a_h, w_h, db2, query_h, query_h, kTD, kD, kFFD, 0, stream);

  // ---- LM head + loss ----
  rms_kernel<<<dim3(kTD), dim3(256), 0, stream>>>(query_h, hn_buf, kTD);
  cvt(hn_buf, a_h, (size_t)kTD * kD, stream);
  cvt(lmW, w_h, (size_t)kV * kD, stream);
  gemm(a_h, w_h, lmb, nullptr, logits, kTD, kV, kD, 0, stream);
  loss_rows_kernel<<<dim3(kTD), dim3(256), 0, stream>>>(logits, targets, nll_buf, val_buf);
  loss_reduce_kernel<<<dim3(1), dim3(256), 0, stream>>>(nll_buf, val_buf, loss);
}
